// MultiHeadAttention_88201448390669
// MI455X (gfx1250) — compile-verified, matrix operand order checked by reading, not run on hardware
//
#include <hip/hip_runtime.h>
#include <hip/hip_bf16.h>

// ---------------------------------------------------------------------------
// MHA forward: B=2, S=2048, DIM=512, H=8, Dh=64 on gfx1250 (wave32, WMMA).
// Kernel 1: QKV projection, 32x64 wave tiles, bf16 WMMA, writes Q,K [bh][s][d]
//           and V transposed [bh][d][s] (bf16) into d_ws.
// Kernel 2: flash attention; 8 waves/block share one (b,h); K/V tiles staged
//           per-block into double-buffered LDS (8x traffic reduction), bf16
//           WMMA for QK^T and PV, fp32 online softmax, fp32 output.
// ---------------------------------------------------------------------------

#define BATCH 2
#define SEQ   2048
#define DIM   512
#define HEADS 8
#define DH    64

typedef __attribute__((ext_vector_type(16))) __bf16   v16bf;
typedef __attribute__((ext_vector_type(2)))  __bf16   v2bf;
typedef __attribute__((ext_vector_type(8)))  float    v8f;
typedef __attribute__((ext_vector_type(8)))  unsigned v8u;

// A-operand (16x32 bf16) K placement for VGPR v, lane-half hf (ISA 7.12.2).
__device__ __forceinline__ int a_kloc(int v, int hf) {
    return (v < 4 ? v * 2 : 16 + (v - 4) * 2) + hf * 8;
}

__device__ __forceinline__ unsigned pack2_bf16(float lo, float hi) {
    v2bf t;
    t[0] = (__bf16)lo;
    t[1] = (__bf16)hi;
    return __builtin_bit_cast(unsigned, t);
}
__device__ __forceinline__ unsigned short f32_to_bf16(float f) {
    __bf16 b = (__bf16)f;
    return __builtin_bit_cast(unsigned short, b);
}

__device__ __forceinline__ v8f wmma_bf16(const unsigned a[8], const unsigned b[8], v8f c) {
    v8u au, bu;
#pragma unroll
    for (int i = 0; i < 8; ++i) { au[i] = a[i]; bu[i] = b[i]; }
    return __builtin_amdgcn_wmma_f32_16x16x32_bf16(
        false, __builtin_bit_cast(v16bf, au),
        false, __builtin_bit_cast(v16bf, bu),
        (short)0, c, false, false);
}

// ---------------------------------------------------------------------------
// Kernel 1: out[m,n] = sum_k X[m,k] * W[n,k] + bias[n]
// One wave computes a 32(m) x 64(n) tile; B-operand regs reused across 2
// m-subtiles (8 WMMAs per 32-wide k-step).
// ---------------------------------------------------------------------------
__global__ __launch_bounds__(256) void qkv_proj_kernel(
    const float* __restrict__ q_in, const float* __restrict__ k_in,
    const float* __restrict__ v_in,
    const float* __restrict__ Wq, const float* __restrict__ bq,
    const float* __restrict__ Wk, const float* __restrict__ bk,
    const float* __restrict__ Wv, const float* __restrict__ bv,
    unsigned short* __restrict__ qb,   // [bh][s][d]
    unsigned short* __restrict__ kb,   // [bh][s][d]
    unsigned short* __restrict__ vt)   // [bh][d][s]
{
    const int wave = blockIdx.x * 8 + (threadIdx.x >> 5);
    const int lane = threadIdx.x & 31;
    const int hf   = lane >> 4;
    const int lr   = lane & 15;

    // 3 mats * (4096/32) m-tiles * (512/64) n-tiles = 3 * 128 * 8 = 3072 waves
    const int mat = wave / 1024;                 // 0=Q 1=K 2=V
    const int rem = wave % 1024;
    const int m0  = (rem >> 3) * 32;
    const int n0  = (rem & 7) * 64;

    const float* X  = (mat == 0) ? q_in : (mat == 1) ? k_in : v_in;
    const float* W  = (mat == 0) ? Wq   : (mat == 1) ? Wk   : Wv;
    const float* Bs = (mat == 0) ? bq   : (mat == 1) ? bk   : bv;

    v8f acc[2][4];
#pragma unroll
    for (int mt = 0; mt < 2; ++mt)
#pragma unroll
        for (int nt = 0; nt < 4; ++nt)
#pragma unroll
            for (int v = 0; v < 8; ++v) acc[mt][nt][v] = 0.0f;

    for (int k0 = 0; k0 < DIM; k0 += 32) {
        unsigned areg[2][8];
#pragma unroll
        for (int mt = 0; mt < 2; ++mt)
#pragma unroll
            for (int v = 0; v < 8; ++v) {
                const int kl = a_kloc(v, hf);
                float2 f = *(const float2*)(X + (size_t)(m0 + mt * 16 + lr) * DIM + k0 + kl);
                areg[mt][v] = pack2_bf16(f.x, f.y);
            }
#pragma unroll
        for (int nt = 0; nt < 4; ++nt) {
            unsigned breg[8];
            const int n = n0 + nt * 16 + lr;   // B-operand: lane holds column n
#pragma unroll
            for (int v = 0; v < 8; ++v) {
                const int kl = a_kloc(v, hf);
                float2 f = *(const float2*)(W + (size_t)n * DIM + k0 + kl);
                breg[v] = pack2_bf16(f.x, f.y);
            }
#pragma unroll
            for (int mt = 0; mt < 2; ++mt)
                acc[mt][nt] = wmma_bf16(areg[mt], breg, acc[mt][nt]);
        }
    }

    // bias + store in head-split bf16 layouts (C layout: row = v + 8*half).
#pragma unroll
    for (int nt = 0; nt < 4; ++nt) {
        const int n    = n0 + nt * 16 + lr;
        const float bb = Bs[n];
        const int h = n >> 6, d = n & 63;
#pragma unroll
        for (int mt = 0; mt < 2; ++mt)
#pragma unroll
            for (int v = 0; v < 8; ++v) {
                const int m = m0 + mt * 16 + v + hf * 8;
                const int b = m >> 11, s = m & (SEQ - 1);
                const int bh = b * HEADS + h;
                const unsigned short bf = f32_to_bf16(acc[mt][nt][v] + bb);
                if (mat == 0)      qb[((size_t)bh * SEQ + s) * DH + d] = bf;
                else if (mat == 1) kb[((size_t)bh * SEQ + s) * DH + d] = bf;
                else               vt[((size_t)bh * DH + d) * SEQ + s] = bf;
            }
    }
}

// ---------------------------------------------------------------------------
// Kernel 2: flash attention. Block = 8 waves, all on the same (b,h); each wave
// owns a 32-row q-tile. K/V 32-key tiles are staged once per block into
// double-buffered LDS and shared by all 8 waves.
// ---------------------------------------------------------------------------
__global__ __launch_bounds__(256) void attn_kernel(
    const unsigned short* __restrict__ qb,
    const unsigned short* __restrict__ kb,
    const unsigned short* __restrict__ vt,
    const unsigned char*  __restrict__ mask,   // [B][S][S] bool
    float* __restrict__ out)                   // [B][S][DIM]
{
    __shared__ unsigned short kbuf[2][32][DH];  // 2 x 4KB  K tiles [key][d]
    __shared__ unsigned short vbuf[2][DH][32];  // 2 x 4KB  V tiles [d][key]
    __shared__ unsigned short pbuf[8][32][32];  // 16KB per-wave P staging

    const int wv   = threadIdx.x >> 5;
    const int lane = threadIdx.x & 31;
    const int hf   = lane >> 4;
    const int lr   = lane & 15;

    const int bh = blockIdx.x >> 3;                       // 0..15, uniform/block
    const int q0 = ((blockIdx.x & 7) * 8 + wv) * 32;      // wave's 32-row q-tile
    const int b  = bh >> 3;
    const int h  = bh & 7;

    const unsigned short* ktile_base = kb + (size_t)bh * SEQ * DH;
    const unsigned short* vtile_base = vt + (size_t)bh * DH * SEQ;

    // Cooperative stage of one 32-key K/V tile into LDS buffer `buf`.
    auto stage = [&](int buf, int jj) {
        {   // K: 32 rows x 64 bf16 = 256 x 16B
            const int row = threadIdx.x >> 3, seg = (threadIdx.x & 7) * 8;
            *(float4*)&kbuf[buf][row][seg] =
                *(const float4*)(ktile_base + (size_t)(jj + row) * DH + seg);
        }
        {   // V: 64 rows x 32 bf16 = 256 x 16B
            const int row = threadIdx.x >> 2, seg = (threadIdx.x & 3) * 8;
            *(float4*)&vbuf[buf][row][seg] =
                *(const float4*)(vtile_base + (size_t)row * SEQ + jj + seg);
        }
    };

    // Q tile (32x64) in A-operand layout: 2 m-subtiles x 2 d-chunks.
    unsigned qa[2][2][8];
#pragma unroll
    for (int mt = 0; mt < 2; ++mt) {
        const unsigned* qrow =
            (const unsigned*)(qb + ((size_t)bh * SEQ + q0 + mt * 16 + lr) * DH);
#pragma unroll
        for (int c = 0; c < 2; ++c)
#pragma unroll
            for (int v = 0; v < 8; ++v)
                qa[mt][c][v] = qrow[(c * 32 + a_kloc(v, hf)) >> 1];
    }

    v8f ctx[2][4];
    float row_m[2][8], row_l[2][8];
#pragma unroll
    for (int mt = 0; mt < 2; ++mt) {
#pragma unroll
        for (int nt = 0; nt < 4; ++nt)
#pragma unroll
            for (int v = 0; v < 8; ++v) ctx[mt][nt][v] = 0.0f;
#pragma unroll
        for (int v = 0; v < 8; ++v) { row_m[mt][v] = -3.0e38f; row_l[mt][v] = 0.0f; }
    }

    const unsigned char* mbase = mask + (size_t)b * SEQ * SEQ;
    const float scale = 0.044194173824159216f;   // 1/sqrt(512)

    stage(0, 0);

    for (int j = 0; j < SEQ; j += 32) {
        const int cur = (j >> 5) & 1;
        __syncthreads();                       // buf[cur] staged, buf[cur^1] free
        if (j + 32 < SEQ) stage(cur ^ 1, j + 32);
        if (j + 64 < SEQ)
            __builtin_prefetch(ktile_base + (size_t)(j + 64 + (threadIdx.x >> 3)) * DH, 0, 1);

        // ---- scores: S = Q K^T (32 x 32); K operands read from LDS ----
        v8f sc[2][2];
#pragma unroll
        for (int mt = 0; mt < 2; ++mt)
#pragma unroll
            for (int nt = 0; nt < 2; ++nt)
#pragma unroll
                for (int v = 0; v < 8; ++v) sc[mt][nt][v] = 0.0f;
#pragma unroll
        for (int nt = 0; nt < 2; ++nt) {
            unsigned kreg[2][8];
#pragma unroll
            for (int c = 0; c < 2; ++c)
#pragma unroll
                for (int v = 0; v < 8; ++v)
                    kreg[c][v] = *(const unsigned*)&kbuf[cur][nt * 16 + lr][c * 32 + a_kloc(v, hf)];
#pragma unroll
            for (int mt = 0; mt < 2; ++mt) {
                sc[mt][nt] = wmma_bf16(qa[mt][0], kreg[0], sc[mt][nt]);
                sc[mt][nt] = wmma_bf16(qa[mt][1], kreg[1], sc[mt][nt]);
            }
        }

        // ---- mask then scale (matches reference order) ----
#pragma unroll
        for (int mt = 0; mt < 2; ++mt)
#pragma unroll
            for (int nt = 0; nt < 2; ++nt)
#pragma unroll
                for (int v = 0; v < 8; ++v) {
                    const int rrow = q0 + mt * 16 + v + hf * 8;
                    const int key  = j + nt * 16 + lr;
                    const unsigned char mm = mbase[(size_t)rrow * SEQ + key];
                    const float val = mm ? sc[mt][nt][v] : -10000.0f;
                    sc[mt][nt][v] = val * scale;
                }

        // ---- online softmax (rows map to acc element v, +8*half) ----
        float p[2][2][8], alpha[2][8];
#pragma unroll
        for (int mt = 0; mt < 2; ++mt)
#pragma unroll
            for (int v = 0; v < 8; ++v) {
                float t = fmaxf(sc[mt][0][v], sc[mt][1][v]);
                t = fmaxf(t, __shfl_xor(t, 1));
                t = fmaxf(t, __shfl_xor(t, 2));
                t = fmaxf(t, __shfl_xor(t, 4));
                t = fmaxf(t, __shfl_xor(t, 8));
                const float mn = fmaxf(row_m[mt][v], t);
                const float a  = __expf(row_m[mt][v] - mn);
                alpha[mt][v] = a;
                row_m[mt][v] = mn;
                const float e0 = __expf(sc[mt][0][v] - mn);
                const float e1 = __expf(sc[mt][1][v] - mn);
                p[mt][0][v] = e0; p[mt][1][v] = e1;
                float rs = e0 + e1;
                rs += __shfl_xor(rs, 1);
                rs += __shfl_xor(rs, 2);
                rs += __shfl_xor(rs, 4);
                rs += __shfl_xor(rs, 8);
                row_l[mt][v] = row_l[mt][v] * a + rs;
            }
#pragma unroll
        for (int mt = 0; mt < 2; ++mt)
#pragma unroll
            for (int nt = 0; nt < 4; ++nt)
#pragma unroll
                for (int v = 0; v < 8; ++v) ctx[mt][nt][v] *= alpha[mt][v];

        // ---- C-layout -> A-layout for P via per-wave LDS tile ----
#pragma unroll
        for (int mt = 0; mt < 2; ++mt)
#pragma unroll
            for (int v = 0; v < 8; ++v) {
                pbuf[wv][mt * 16 + v + hf * 8][lr]      = f32_to_bf16(p[mt][0][v]);
                pbuf[wv][mt * 16 + v + hf * 8][16 + lr] = f32_to_bf16(p[mt][1][v]);
            }
        asm volatile("" ::: "memory");   // same-wave LDS is in-order in HW
        unsigned pa[2][8];
#pragma unroll
        for (int mt = 0; mt < 2; ++mt)
#pragma unroll
            for (int v = 0; v < 8; ++v)
                pa[mt][v] = *(const unsigned*)&pbuf[wv][mt * 16 + lr][a_kloc(v, hf)];
        asm volatile("" ::: "memory");

        // ---- ctx += P V ; V operands read from LDS, shared by m-subtiles ----
#pragma unroll
        for (int nt = 0; nt < 4; ++nt) {
            unsigned vreg[8];
#pragma unroll
            for (int v = 0; v < 8; ++v)
                vreg[v] = *(const unsigned*)&vbuf[cur][nt * 16 + lr][a_kloc(v, hf)];
#pragma unroll
            for (int mt = 0; mt < 2; ++mt)
                ctx[mt][nt] = wmma_bf16(pa[mt], vreg, ctx[mt][nt]);
        }
    }

    // ---- normalize + merge heads: out[b][s][h*64+d] fp32 ----
#pragma unroll
    for (int mt = 0; mt < 2; ++mt)
#pragma unroll
        for (int nt = 0; nt < 4; ++nt)
#pragma unroll
            for (int v = 0; v < 8; ++v) {
                const int rrow = q0 + mt * 16 + v + hf * 8;
                const int col  = h * DH + nt * 16 + lr;
                out[((size_t)b * SEQ + rrow) * DIM + col] = ctx[mt][nt][v] / row_l[mt][v];
            }
}

extern "C" void kernel_launch(void* const* d_in, const int* in_sizes, int n_in,
                              void* d_out, int out_size, void* d_ws, size_t ws_size,
                              hipStream_t stream) {
    (void)in_sizes; (void)n_in; (void)out_size; (void)ws_size;
    const float* q  = (const float*)d_in[0];
    const float* k  = (const float*)d_in[1];
    const float* v  = (const float*)d_in[2];
    const unsigned char* mask = (const unsigned char*)d_in[3];
    const float* Wq = (const float*)d_in[4];
    const float* bq = (const float*)d_in[5];
    const float* Wk = (const float*)d_in[6];
    const float* bk = (const float*)d_in[7];
    const float* Wv = (const float*)d_in[8];
    const float* bv = (const float*)d_in[9];

    const size_t elems = (size_t)BATCH * HEADS * SEQ * DH;  // 2,097,152 per tensor
    unsigned short* qb = (unsigned short*)d_ws;
    unsigned short* kb = qb + elems;
    unsigned short* vt = kb + elems;
    float* out = (float*)d_out;

    // 3 mats * 128 m-tiles * 8 n-tiles = 3072 waves -> 384 blocks of 8 waves
    qkv_proj_kernel<<<384, 256, 0, stream>>>(q, k, v, Wq, bq, Wk, bk, Wv, bv,
                                             qb, kb, vt);
    // 16 (b,h) * 8 q-blocks = 128 blocks of 8 waves (one 32-row q-tile each)
    attn_kernel<<<128, 256, 0, stream>>>(qb, kb, vt, mask, out);
}
